// UpdateM_12094627906028
// MI455X (gfx1250) — matrix operand order repaired and checked by reading, not compiled, not run on hardware
//
#include <hip/hip_runtime.h>
#include <hip/hip_bf16.h>
#include <stdint.h>

// Shapes from the reference
#define LL   384          // L
#define RR   256          // R
#define IND  128          // IN_DIM == PAIR_DIM
#define NH   8            // N_HEADS
#define HD   16           // HEAD_DIM
#define FFH  512          // IN_DIM * FF_MULT
#define NEIN (RR*HD)      // 4096, N dim of the einsum GEMM

typedef _Float16 half_t;
typedef __attribute__((ext_vector_type(16))) _Float16 v16h;
typedef __attribute__((ext_vector_type(8)))  float    v8f;
typedef __attribute__((ext_vector_type(4)))  float    f32x4;
typedef __attribute__((ext_vector_type(2)))  float    f32x2;

union AFrag { v16h v; f32x4 f[2]; };
union H4    { half_t h[4]; f32x2 f; };

// ---------------- wave32 reductions ----------------
__device__ __forceinline__ float wsum(float v) {
  for (int o = 16; o; o >>= 1) v += __shfl_xor(v, o, 32);
  return v;
}
__device__ __forceinline__ float wmaxr(float v) {
  for (int o = 16; o; o >>= 1) v = fmaxf(v, __shfl_xor(v, o, 32));
  return v;
}

// ---------------- WMMA fragment loaders (ISA 7.12.2 layouts) ----------------
// A 16x32 f16: lane l holds row m0+(l&15); halves 0..7 at K=k0+(l>>4)*8,
// halves 8..15 at K=k0+16+(l>>4)*8. Requires k0,lda multiples of 8 (16B align).
__device__ __forceinline__ v16h load_a16x32(const half_t* A, int lda, int m0, int k0) {
  const int lane = threadIdx.x & 31;
  const half_t* p = A + (size_t)(m0 + (lane & 15)) * lda + k0 + (lane >> 4) * 8;
  AFrag u;
  u.f[0] = *(const f32x4*)p;
  u.f[1] = *(const f32x4*)(p + 16);
  return u.v;
}
// B 32x16 f16 (row-major KxN): lane l holds column n0+(l&15); halves 0..7 at
// K=k0+(l>>4)*8, halves 8..15 at K=k0+16+(l>>4)*8.
__device__ __forceinline__ v16h load_b32x16(const half_t* B, int ldb, int k0, int n0) {
  const int lane = threadIdx.x & 31;
  const half_t* p = B + (size_t)(k0 + (lane >> 4) * 8) * ldb + n0 + (lane & 15);
  v16h v;
#pragma unroll
  for (int s = 0; s < 8; ++s) {
    v[s] = p[(size_t)s * ldb];
    v[s + 8] = p[(size_t)(s + 16) * ldb];
  }
  return v;
}
__device__ __forceinline__ v8f wmma_f16(v16h a, v16h b, v8f c) {
  return __builtin_amdgcn_wmma_f32_16x16x32_f16(false, a, false, b, (short)0, c, false, false);
}

// ---- CDNA5 async copy: global -> LDS, tracked by ASYNCcnt (ISA 15.18.3) ----
__device__ __forceinline__ void async_g2l_b128(uint32_t lds_off, const void* gaddr) {
  const uint64_t ga = (uint64_t)(uintptr_t)gaddr;
  asm volatile("global_load_async_to_lds_b128 %0, %1, off"
               :: "v"(lds_off), "v"(ga) : "memory");
}
__device__ __forceinline__ void wait_asynccnt0() {
  asm volatile("s_wait_asynccnt 0x0" ::: "memory");
}
// Ordering anchor: ties an *integer* slab offset through a VGPR so every
// derived LDS address is data-dependent on a point after the asynccnt wait +
// barrier (the async LDS writes are invisible to alias analysis, which would
// otherwise hoist the slab reads above the sync). Anchoring the integer, not
// the pointer, keeps the base addrspace(3)-derived so the reads stay ds_load.
__device__ __forceinline__ uint32_t anchor_u32(uint32_t v) {
  asm volatile("" : "+v"(v));
  return v;
}

// ---------------- K0: fp32 -> fp16 weight conversion ----------------
__global__ void k_cvt(const float* __restrict__ fw1, const float* __restrict__ fw2,
                      half_t* __restrict__ w1h, half_t* __restrict__ w2h) {
  const int idx = blockIdx.x * 256 + threadIdx.x;   // 65536 threads
  w1h[idx] = (half_t)fw1[idx];
  w2h[idx] = (half_t)fw2[idx];
}

// ---------------- K1: symmetrize + LN1 + logits + softmax(j) ----------------
__global__ void k_attn(const float* __restrict__ x,
                       const float* __restrict__ g1, const float* __restrict__ be1,
                       const float* __restrict__ w1, const float* __restrict__ b1,
                       half_t* __restrict__ attn) {
  __shared__ float lg[LL][NH];
  const int i = blockIdx.x;
  const int tid = threadIdx.x, wid = tid >> 5, lane = tid & 31;
  const int d0 = lane * 4;

  float gg[4], bb[4], w1r[4][NH];
#pragma unroll
  for (int q = 0; q < 4; ++q) {
    gg[q] = g1[d0 + q]; bb[q] = be1[d0 + q];
#pragma unroll
    for (int h = 0; h < NH; ++h) w1r[q][h] = w1[(d0 + q) * NH + h];
  }

  for (int j = wid; j < LL; j += 8) {
    const f32x4 a = *(const f32x4*)(x + ((size_t)i * LL + j) * IND + d0);
    const f32x4 b = *(const f32x4*)(x + ((size_t)j * LL + i) * IND + d0);
    float p[4] = {0.5f * (a.x + b.x), 0.5f * (a.y + b.y),
                  0.5f * (a.z + b.z), 0.5f * (a.w + b.w)};
    float mu = wsum(p[0] + p[1] + p[2] + p[3]) * (1.0f / IND);
    float c[4], vv = 0.f;
#pragma unroll
    for (int q = 0; q < 4; ++q) { c[q] = p[q] - mu; vv += c[q] * c[q]; }
    const float inv = rsqrtf(wsum(vv) * (1.0f / IND) + 1e-5f);
    float n[4];
#pragma unroll
    for (int q = 0; q < 4; ++q) n[q] = c[q] * inv * gg[q] + bb[q];
    float lp[NH];
#pragma unroll
    for (int h = 0; h < NH; ++h)
      lp[h] = n[0]*w1r[0][h] + n[1]*w1r[1][h] + n[2]*w1r[2][h] + n[3]*w1r[3][h];
    for (int o = 16; o; o >>= 1) {
#pragma unroll
      for (int h = 0; h < NH; ++h) lp[h] += __shfl_xor(lp[h], o, 32);
    }
    if (lane == 0) {
#pragma unroll
      for (int h = 0; h < NH; ++h) lg[j][h] = lp[h] + b1[h];
    }
  }
  __syncthreads();

  if (wid < NH) {
    const int h = wid;
    float e[LL / 32];
    float mx = -3.4e38f;
#pragma unroll
    for (int t = 0; t < LL / 32; ++t) {
      const float v = lg[lane + 32 * t][h];
      e[t] = v; mx = fmaxf(mx, v);
    }
    mx = wmaxr(mx);
    float sm = 0.f;
#pragma unroll
    for (int t = 0; t < LL / 32; ++t) { const float z = __expf(e[t] - mx); e[t] = z; sm += z; }
    const float r = 1.0f / wsum(sm);
    half_t* dst = attn + ((size_t)h * LL + i) * LL;
#pragma unroll
    for (int t = 0; t < LL / 32; ++t) dst[lane + 32 * t] = (half_t)(e[t] * r);
  }
}

// ---------------- K2: values = LN2(m) @ w2 + b2 -> V[j][r*16+d] f16 ----------
__global__ void k_values(const float* __restrict__ m,
                         const float* __restrict__ g2, const float* __restrict__ be2,
                         const float* __restrict__ w2, const float* __restrict__ b2,
                         half_t* __restrict__ V) {
  const int rid = blockIdx.x * 8 + (threadIdx.x >> 5);   // r*384 + j
  const int lane = threadIdx.x & 31, d0 = lane * 4;
  const int r = rid / LL, j = rid % LL;

  const f32x4 mv = *(const f32x4*)(m + (size_t)rid * IND + d0);
  float p[4] = {mv.x, mv.y, mv.z, mv.w};
  const float mu = wsum(p[0] + p[1] + p[2] + p[3]) * (1.0f / IND);
  float c[4], vv = 0.f;
#pragma unroll
  for (int q = 0; q < 4; ++q) { c[q] = p[q] - mu; vv += c[q] * c[q]; }
  const float inv = rsqrtf(wsum(vv) * (1.0f / IND) + 1e-5f);
  float n[4];
#pragma unroll
  for (int q = 0; q < 4; ++q) n[q] = c[q] * inv * g2[d0 + q] + be2[d0 + q];

  float acc[HD];
#pragma unroll
  for (int dd = 0; dd < HD; ++dd) acc[dd] = 0.f;
#pragma unroll
  for (int q = 0; q < 4; ++q) {
    const float* wr = w2 + (size_t)(d0 + q) * HD;
#pragma unroll
    for (int dd = 0; dd < HD; ++dd) acc[dd] += n[q] * wr[dd];
  }
  for (int o = 16; o; o >>= 1) {
#pragma unroll
    for (int dd = 0; dd < HD; ++dd) acc[dd] += __shfl_xor(acc[dd], o, 32);
  }
  if (lane == 0) {
    AFrag u;
    half_t* hh = (half_t*)&u.v;
#pragma unroll
    for (int dd = 0; dd < HD; ++dd) hh[dd] = (half_t)(acc[dd] + b2[dd]);
    half_t* dst = V + (size_t)j * NEIN + r * HD;
    *(f32x4*)dst = u.f[0];
    *(f32x4*)(dst + 8) = u.f[1];
  }
}

// ---- K3: per-head GEMM A_h[384x384] @ V[384x4096], + m ---------------------
// Block = 128x64 tile (8 waves x 16 rows), B slab (32x64) staged in LDS via
// global_load_async_to_lds_b128, double-buffered, ASYNCcnt + barrier synced.
#define BSTRIDE 72      // 64 + 8 halfs pad: 144 B rows, 16B-aligned, bank-spread
__global__ void k_attn_gemm(const half_t* __restrict__ attn, const half_t* __restrict__ V,
                            const float* __restrict__ m, float* __restrict__ outb) {
  __shared__ half_t bs[2][32 * BSTRIDE];     // 2 x 4608 B slabs
  const int tid = threadIdx.x, wid = tid >> 5, lane = tid & 31;
  const int MB = LL / 128, NB = NEIN / 64;                 // 3, 64
  const int h = blockIdx.x / (MB * NB);
  const int rem = blockIdx.x % (MB * NB);
  const int i0 = (rem / NB) * 128 + wid * 16;
  const int n0 = (rem % NB) * 64;
  const half_t* A = attn + (size_t)h * LL * LL;

  // this thread's async chunk: row = tid>>3 (0..31), cc = tid&7 (16B chunks)
  const int arow = tid >> 3, acc8 = tid & 7;
  const half_t* gsrc = V + (size_t)arow * NEIN + n0 + acc8 * 8;
  const uint32_t ldsc = (uint32_t)(uintptr_t)(&bs[0][0]) +
                        (uint32_t)(arow * BSTRIDE + acc8 * 8) * 2u;
  const uint32_t bufStep = 32 * BSTRIDE * 2u;

  async_g2l_b128(ldsc, gsrc);                               // k-step 0 slab

  v8f z = {};
  v8f acc[4] = {z, z, z, z};
  const int NK = LL / 32;                                   // 12
  for (int ks = 0; ks < NK; ++ks) {
    wait_asynccnt0();          // own chunks of slab ks have landed in LDS
    __syncthreads();           // all waves landed + done consuming buf (ks+1)&1
    // integer anchor AFTER the sync point: slab reads depend on it (ds path kept)
    const uint32_t bsel = anchor_u32((uint32_t)(ks & 1) * (32u * BSTRIDE));
    const half_t* bsl = &bs[0][0] + bsel;
    if (ks + 1 < NK)
      async_g2l_b128(ldsc + ((ks + 1) & 1) * bufStep,
                     gsrc + (size_t)(ks + 1) * 32 * NEIN);
    const int k0 = ks * 32;
    const v16h aF = load_a16x32(A, LL, i0, k0);
    if (k0 + 32 < LL)
      __builtin_prefetch(A + (size_t)i0 * LL + k0 + 32, 0, 0);  // global_prefetch_b8
#pragma unroll
    for (int q = 0; q < 4; ++q) {
      const v16h bF = load_b32x16(bsl, BSTRIDE, 0, 16 * q);
      acc[q] = wmma_f16(aF, bF, acc[q]);
    }
  }

  const int hi = lane >> 4, nl = lane & 15;
#pragma unroll
  for (int q = 0; q < 4; ++q) {
    const int col = n0 + 16 * q + nl;      // col = r*16 + d
    const int r = col >> 4, dd = col & 15;
#pragma unroll
    for (int v = 0; v < 8; ++v) {
      const int i = i0 + v + 8 * hi;
      const size_t idx = ((size_t)r * LL + i) * IND + h * HD + dd;
      outb[idx] = m[idx] + acc[q][v];      // out = m + attn_out
    }
  }
}

// ---------------- K4: residue = LN3(out) -> f16 ----------------
__global__ void k_ln3(const float* __restrict__ outb,
                      const float* __restrict__ g3, const float* __restrict__ be3,
                      half_t* __restrict__ res) {
  const int rid = blockIdx.x * 8 + (threadIdx.x >> 5);
  const int lane = threadIdx.x & 31, d0 = lane * 4;
  const f32x4 o = *(const f32x4*)(outb + (size_t)rid * IND + d0);
  float p[4] = {o.x, o.y, o.z, o.w};
  const float mu = wsum(p[0] + p[1] + p[2] + p[3]) * (1.0f / IND);
  float c[4], vv = 0.f;
#pragma unroll
  for (int q = 0; q < 4; ++q) { c[q] = p[q] - mu; vv += c[q] * c[q]; }
  const float inv = rsqrtf(wsum(vv) * (1.0f / IND) + 1e-5f);
  H4 u;
#pragma unroll
  for (int q = 0; q < 4; ++q) u.h[q] = (half_t)(c[q] * inv * g3[d0 + q] + be3[d0 + q]);
  *(f32x2*)(res + (size_t)rid * IND + d0) = u.f;
}

// ---------------- K5: fused FFN: out += relu(res@W1+b1)@W2+b2 --------------
#define H1_STRIDE 520   // 512 + 8 halfs pad: keeps 16B alignment, spreads banks
__global__ void k_ff(const half_t* __restrict__ res, const half_t* __restrict__ w1h,
                     const half_t* __restrict__ w2h, const float* __restrict__ fb1,
                     const float* __restrict__ fb2, const float* __restrict__ outb,
                     float* __restrict__ out) {
  extern __shared__ char smem[];
  half_t* h1 = (half_t*)smem;                       // [64][520] f16 hidden tile
  const int tid = threadIdx.x, wid = tid >> 5, lane = tid & 31;
  const int rowBase = blockIdx.x * 64;
  const int hi = lane >> 4, nl = lane & 15;
  const int wm0 = (wid & 3) * 16;

  // Phase 1: H = relu(residue[64x128] @ ffw1[128x512] + b1) -> LDS (f16)
  {
    const int wn0 = (wid >> 2) * 256;
    v8f z = {};
    v8f acc[16];
#pragma unroll
    for (int q = 0; q < 16; ++q) acc[q] = z;
    for (int k0 = 0; k0 < IND; k0 += 32) {
      const v16h aF = load_a16x32(res, IND, rowBase + wm0, k0);
#pragma unroll
      for (int q = 0; q < 16; ++q) {
        const v16h bF = load_b32x16(w1h, FFH, k0, wn0 + 16 * q);
        acc[q] = wmma_f16(aF, bF, acc[q]);
      }
    }
#pragma unroll
    for (int q = 0; q < 16; ++q) {
      const int col = wn0 + 16 * q + nl;
      const float bias = fb1[col];
#pragma unroll
      for (int v = 0; v < 8; ++v) {
        const int row = wm0 + v + 8 * hi;
        h1[row * H1_STRIDE + col] = (half_t)fmaxf(acc[q][v] + bias, 0.0f);
      }
    }
  }
  __syncthreads();

  // Phase 2: out = outb + H[64x512] @ ffw2[512x128] + b2
  {
    const int wn0 = (wid >> 2) * 64;
    v8f z = {};
    v8f acc[4] = {z, z, z, z};
    for (int k0 = 0; k0 < FFH; k0 += 32) {
      const v16h aF = load_a16x32(h1, H1_STRIDE, wm0, k0);   // from LDS (transposed path)
#pragma unroll
      for (int q = 0; q < 4; ++q) {
        const v16h bF = load_b32x16(w2h, IND, k0, wn0 + 16 * q);
        acc[q] = wmma_f16(aF, bF, acc[q]);
      }
    }
#pragma unroll
    for (int q = 0; q < 4; ++q) {
      const int col = wn0 + 16 * q + nl;
      const float bias = fb2[col];
#pragma unroll
      for (int v = 0; v < 8; ++v) {
        const int grow = rowBase + wm0 + v + 8 * hi;
        const size_t idx = (size_t)grow * IND + col;
        out[idx] = outb[idx] + acc[q][v] + bias;
      }
    }
  }
}

// ---------------- launcher ----------------
extern "C" void kernel_launch(void* const* d_in, const int* in_sizes, int n_in,
                              void* d_out, int out_size, void* d_ws, size_t ws_size,
                              hipStream_t stream) {
  (void)in_sizes; (void)n_in; (void)out_size; (void)ws_size;
  const float* x   = (const float*)d_in[0];
  const float* m   = (const float*)d_in[1];
  const float* g1  = (const float*)d_in[2];
  const float* be1 = (const float*)d_in[3];
  const float* g2  = (const float*)d_in[4];
  const float* be2 = (const float*)d_in[5];
  const float* g3  = (const float*)d_in[6];
  const float* be3 = (const float*)d_in[7];
  const float* w1  = (const float*)d_in[8];
  const float* b1  = (const float*)d_in[9];
  const float* w2  = (const float*)d_in[10];
  const float* b2  = (const float*)d_in[11];
  const float* fw1 = (const float*)d_in[12];
  const float* fb1 = (const float*)d_in[13];
  const float* fw2 = (const float*)d_in[14];
  const float* fb2 = (const float*)d_in[15];
  float* out = (float*)d_out;

  // workspace carve-up (all offsets 256B-aligned)
  char* ws = (char*)d_ws;
  size_t o = 0;
  half_t* attn = (half_t*)(ws + o); o += (size_t)NH * LL * LL * 2;      //  2.36 MB
  half_t* V    = (half_t*)(ws + o); o += (size_t)LL * NEIN * 2;         //  3.15 MB
  float*  outb = (float*) (ws + o); o += (size_t)RR * LL * IND * 4;     // 50.3  MB
  half_t* res  = (half_t*)(ws + o); o += (size_t)RR * LL * IND * 2;     // 25.2  MB
  half_t* w1h  = (half_t*)(ws + o); o += (size_t)IND * FFH * 2;         //  0.13 MB
  half_t* w2h  = (half_t*)(ws + o); o += (size_t)FFH * IND * 2;         //  0.13 MB

  k_cvt   <<<256, 256, 0, stream>>>(fw1, fw2, w1h, w2h);
  k_attn  <<<LL, 256, 0, stream>>>(x, g1, be1, w1, b1, attn);
  k_values<<<(RR * LL) / 8, 256, 0, stream>>>(m, g2, be2, w2, b2, V);
  k_attn_gemm<<<NH * (LL / 128) * (NEIN / 64), 256, 0, stream>>>(attn, V, m, outb);
  k_ln3   <<<(RR * LL) / 8, 256, 0, stream>>>(outb, g3, be3, res);
  const size_t smem = (size_t)64 * H1_STRIDE * sizeof(half_t);          // 66,560 B
  k_ff    <<<(RR * LL) / 64, 256, smem, stream>>>(res, w1h, w2h, fb1, fb2, outb, out);
}
